// Block_83700322664718
// MI455X (gfx1250) — matrix operand order repaired and checked by reading, not compiled, not run on hardware
//
#include <hip/hip_runtime.h>
#include <hip/hip_bf16.h>
#include <math.h>

// GPT-2 transformer block, SEQ=4096, d=768, 12 heads x 64.
// Compute/L2-bound -> all GEMM-shaped work on v_wmma_f32_16x16x32_bf16
// (f32 accumulate); GEMM K-slices staged to LDS via async global->LDS copies
// (ASYNCcnt) with double buffering.

#define SEQL   4096
#define DMODEL 768
#define NHEAD  12
#define HDIM   64
#define QKVD   2304   // 3*DMODEL, torch order: k | q | v

typedef __attribute__((ext_vector_type(16))) __bf16 v16bf;
typedef __attribute__((ext_vector_type(8)))  float  v8f;

__device__ __forceinline__ unsigned short f2bf(float x) {
  union { float f; unsigned int u; } c; c.f = x;
  unsigned int r = c.u + 0x7FFFu + ((c.u >> 16) & 1u);   // round-to-nearest-even
  return (unsigned short)(r >> 16);
}

// Build a 16xbf16 fragment from two 16-byte chunks (global or LDS).
__device__ __forceinline__ v16bf load_frag(const unsigned short* p0,
                                           const unsigned short* p1) {
  union { uint4 u[2]; v16bf v; } f;
  f.u[0] = *(const uint4*)p0;
  f.u[1] = *(const uint4*)p1;
  return f.v;
}

__device__ __forceinline__ v8f wmma_bf16(v16bf a, v16bf b, v8f c) {
  return __builtin_amdgcn_wmma_f32_16x16x32_bf16(false, a, false, b,
                                                 (short)0, c, false, false);
}

// ---- CDNA5 async global->LDS copy (ASYNCcnt), 16B per lane ------------------
// lds_off: byte offset in LDS (generic-shared addr truncated to 32b = LDS addr)
__device__ __forceinline__ void async_copy_b128(unsigned lds_off,
                                                const unsigned short* g) {
  asm volatile("global_load_async_to_lds_b128 %0, %1, off"
               :: "v"(lds_off), "v"(g) : "memory");
}
__device__ __forceinline__ void wait_async0() {
#if defined(__has_builtin) && __has_builtin(__builtin_amdgcn_s_wait_asynccnt)
  __builtin_amdgcn_s_wait_asynccnt(0);
#else
  asm volatile("s_wait_asynccnt 0x0" ::: "memory");
#endif
}
__device__ __forceinline__ unsigned lds_addr(const void* p) {
  return (unsigned)(size_t)p;   // low 32 bits of generic LDS addr = LDS offset
}

// ---------------------------------------------------------------- weights
// in: K x N row-major f32   ->   out: N x K row-major bf16 (i.e. W^T)
__global__ void transpose_to_bf16(const float* __restrict__ in,
                                  unsigned short* __restrict__ out,
                                  int K, int N) {
  long long t = (long long)blockIdx.x * 256 + threadIdx.x;
  if (t >= (long long)K * N) return;
  int k = (int)(t / N), n = (int)(t % N);
  out[(size_t)n * K + k] = f2bf(in[t]);
}

// ---------------------------------------------------------------- layernorm
__global__ __launch_bounds__(256)
void layernorm_bf16(const float* __restrict__ x, const float* __restrict__ g,
                    const float* __restrict__ b, unsigned short* __restrict__ y,
                    int D) {
  __shared__ float red[256];
  const int tid = threadIdx.x;
  const float* xr = x + (size_t)blockIdx.x * D;
  float s = 0.f;
  for (int i = tid; i < D; i += 256) s += xr[i];
  red[tid] = s; __syncthreads();
  for (int o = 128; o > 0; o >>= 1) { if (tid < o) red[tid] += red[tid + o]; __syncthreads(); }
  const float mu = red[0] / (float)D;
  __syncthreads();
  float s2 = 0.f;
  for (int i = tid; i < D; i += 256) { float d = xr[i] - mu; s2 += d * d; }
  red[tid] = s2; __syncthreads();
  for (int o = 128; o > 0; o >>= 1) { if (tid < o) red[tid] += red[tid + o]; __syncthreads(); }
  const float rstd = rsqrtf(red[0] / (float)D + 1e-5f);
  unsigned short* yr = y + (size_t)blockIdx.x * D;
  for (int i = tid; i < D; i += 256)
    yr[i] = f2bf((xr[i] - mu) * rstd * g[i] + b[i]);
}

// ---------------------------------------------------------------- GEMM
// C[M][N] = A[M][K] (bf16 row-major) * B (given as BT[N][K] bf16 row-major)
// EPI: 0 = bias -> f32 | 1 = bias + residual -> f32
//      2 = bias + exact GELU -> bf16 | 3 = bias -> bf16
// 128 threads = 4 waves; wave tile 64x64; block tile 128x128.
// K-slices (128x32 of A and of BT) double-buffered in LDS via async copies.

// issue async copies for one 32-wide K slice (8KB A + 8KB B); thread t owns row t
__device__ __forceinline__ void gemm_stage(const unsigned short* A,
                                           const unsigned short* BT,
                                           int Mb, int Nb, int K, int k0,
                                           unsigned short* Albs,
                                           unsigned short* Blbs, int t) {
  const unsigned short* ga = A  + (size_t)(Mb + t) * K + k0;
  const unsigned short* gb = BT + (size_t)(Nb + t) * K + k0;
  const unsigned la = lds_addr(Albs + t * 32);
  const unsigned lb = lds_addr(Blbs + t * 32);
#pragma unroll
  for (int c = 0; c < 4; ++c) {
    async_copy_b128(la + c * 16, ga + c * 8);
    async_copy_b128(lb + c * 16, gb + c * 8);
  }
}

template <int EPI>
__global__ __launch_bounds__(128)
void gemm_bf16_wmma(const unsigned short* __restrict__ A,
                    const unsigned short* __restrict__ BT,
                    const float* __restrict__ bias,
                    const float* __restrict__ residual,
                    float* __restrict__ outF, unsigned short* __restrict__ outB,
                    int M, int N, int K) {
  __shared__ __align__(16) unsigned short Ald[2][128 * 32];
  __shared__ __align__(16) unsigned short Bld[2][128 * 32];
  const int tid  = threadIdx.x;
  const int lane = tid & 31;
  const int wave = tid >> 5;
  const int hx   = lane >> 4;          // half-wave (0/1)
  const int ln16 = lane & 15;
  const int Mb = blockIdx.y * 128;
  const int Nb = blockIdx.x * 128;
  const int wr = (wave >> 1) * 64;     // wave row offset in block tile
  const int wc = (wave & 1) * 64;      // wave col offset in block tile
  (void)M;

  const v8f vzero = {0.f,0.f,0.f,0.f,0.f,0.f,0.f,0.f};
  v8f acc[4][4];
#pragma unroll
  for (int i = 0; i < 4; ++i)
#pragma unroll
    for (int j = 0; j < 4; ++j) acc[i][j] = vzero;

  gemm_stage(A, BT, Mb, Nb, K, 0, &Ald[0][0], &Bld[0][0], tid);

  for (int k0 = 0; k0 < K; k0 += 32) {
    const int buf = (k0 >> 5) & 1;
    wait_async0();        // this wave's copies done
    __syncthreads();      // -> all waves' copies into buf done
    if (k0 + 32 < K)      // prefetch next slice into the other buffer
      gemm_stage(A, BT, Mb, Nb, K, k0 + 32, &Ald[buf ^ 1][0], &Bld[buf ^ 1][0], tid);

    v16bf a[4], b[4];
#pragma unroll
    for (int i = 0; i < 4; ++i) {   // A frag: local row wr+16i+ln16, lda=32
      const unsigned short* p = &Ald[buf][(wr + 16 * i + ln16) * 32 + hx * 8];
      a[i] = load_frag(p, p + 16);
    }
#pragma unroll
    for (int j = 0; j < 4; ++j) {   // B frag: local col wc+16j+ln16, ldb=32
      const unsigned short* p = &Bld[buf][(wc + 16 * j + ln16) * 32 + hx * 8];
      b[j] = load_frag(p, p + 16);
    }
#pragma unroll
    for (int i = 0; i < 4; ++i)
#pragma unroll
      for (int j = 0; j < 4; ++j)
        acc[i][j] = wmma_bf16(a[i], b[j], acc[i][j]);
  }

#pragma unroll
  for (int i = 0; i < 4; ++i)
#pragma unroll
    for (int j = 0; j < 4; ++j) {
      const int col = Nb + wc + 16 * j + ln16;
      const float bv = bias[col];
#pragma unroll
      for (int v = 0; v < 8; ++v) {   // C layout: lane holds col, rows v+8*hx
        const int row = Mb + wr + 16 * i + v + 8 * hx;
        float val = acc[i][j][v] + bv;
        const size_t idx = (size_t)row * N + col;
        if (EPI == 0)      outF[idx] = val;
        else if (EPI == 1) outF[idx] = val + residual[idx];
        else if (EPI == 2) outB[idx] = f2bf(0.5f * val * (1.f + erff(val * 0.70710678118f)));
        else               outB[idx] = f2bf(val);
      }
    }
}

// ---------------------------------------------------------------- attention
// qkv: [SEQ][2304] bf16 (k@0, q@768, v@1536; head h -> cols h*64..h*64+63)
// out: [SEQ][768] bf16. Grid: (SEQ/64, NHEAD), 128 threads (4 waves).
// Each wave owns 16 query rows; flash-style online softmax.
__global__ __launch_bounds__(128)
void attn_wmma(const unsigned short* __restrict__ qkv,
               unsigned short* __restrict__ out) {
  __shared__ __align__(16) unsigned short vt[HDIM * 64];   // V^T: [dim][key]
  __shared__ __align__(16) unsigned short pl[4 * 16 * 64]; // P per wave [row][key]

  const int lane = threadIdx.x & 31;
  const int wave = threadIdx.x >> 5;
  const int hx   = lane >> 4;
  const int ln16 = lane & 15;
  const int h    = blockIdx.y;
  const int qb   = blockIdx.x;
  const int qbase = qb * 64 + wave * 16;

  // Q fragments (A layout), K-dim split 0..31 / 32..63
  v16bf qf[2];
#pragma unroll
  for (int ks = 0; ks < 2; ++ks) {
    const unsigned short* p =
        qkv + (size_t)(qbase + ln16) * QKVD + DMODEL + h * HDIM + ks * 32 + hx * 8;
    qf[ks] = load_frag(p, p + 16);
  }

  const v8f vzero = {0.f,0.f,0.f,0.f,0.f,0.f,0.f,0.f};
  float m[8], l[8];
  v8f o[4];
#pragma unroll
  for (int v = 0; v < 8; ++v) { m[v] = -INFINITY; l[v] = 0.f; }
#pragma unroll
  for (int j = 0; j < 4; ++j) o[j] = vzero;

  for (int kb = 0; kb <= qb; ++kb) {
    // stage V^T into LDS: coalesced 16B global loads + b16 LDS scatter
    for (int t = threadIdx.x; t < 64 * 8; t += 128) {
      const int key = t >> 3, c = t & 7;        // dims c*8 .. c*8+7 of one key
      union { uint4 u; unsigned short s[8]; } d;
      d.u = *(const uint4*)(qkv + (size_t)(kb * 64 + key) * QKVD +
                            2 * DMODEL + h * HDIM + c * 8);
#pragma unroll
      for (int e = 0; e < 8; ++e) vt[(c * 8 + e) * 64 + key] = d.s[e];
    }
    __syncthreads();

    // S = Q K^T : K rows ([key][dim]) are exactly the B^T layout -> direct loads
    v8f s[4];
#pragma unroll
    for (int j = 0; j < 4; ++j) s[j] = vzero;
#pragma unroll
    for (int ks = 0; ks < 2; ++ks)
#pragma unroll
      for (int j = 0; j < 4; ++j) {
        const unsigned short* p =
            qkv + (size_t)(kb * 64 + 16 * j + ln16) * QKVD + h * HDIM + ks * 32 + hx * 8;
        s[j] = wmma_bf16(qf[ks], load_frag(p, p + 16), s[j]);
      }

    // scale + causal mask + online softmax; row stats reduced over the
    // 16-lane half (xor masks 1,2,4,8 keep lane/16 fixed).
#pragma unroll
    for (int v = 0; v < 8; ++v) {
      const int row = qbase + v + 8 * hx;
      float sv[4], rm = -INFINITY;
#pragma unroll
      for (int j = 0; j < 4; ++j) {
        const int key = kb * 64 + 16 * j + ln16;
        float xv = s[j][v] * 0.125f;            // 1/sqrt(64)
        if (key > row) xv = -INFINITY;
        sv[j] = xv;
        rm = fmaxf(rm, xv);
      }
#pragma unroll
      for (int msk = 1; msk < 16; msk <<= 1) rm = fmaxf(rm, __shfl_xor(rm, msk, 32));
      const float nm    = fmaxf(m[v], rm);
      const float scale = expf(m[v] - nm);      // expf(-inf)=0 on first block
      float rs = 0.f;
#pragma unroll
      for (int j = 0; j < 4; ++j) {
        const float pv = expf(sv[j] - nm);      // masked -> exp(-inf)=0
        rs += pv;
        pl[wave * 1024 + (v + 8 * hx) * 64 + 16 * j + ln16] = f2bf(pv);
      }
#pragma unroll
      for (int msk = 1; msk < 16; msk <<= 1) rs += __shfl_xor(rs, msk, 32);
      l[v] = l[v] * scale + rs;
      m[v] = nm;
#pragma unroll
      for (int j = 0; j < 4; ++j) o[j][v] = o[j][v] * scale;
    }

    // O += P * V : P re-read from LDS in A layout, V^T in LDS gives B layout
#pragma unroll
    for (int ks = 0; ks < 2; ++ks) {
      const unsigned short* ap = &pl[wave * 1024 + ln16 * 64 + ks * 32 + hx * 8];
      const v16bf af = load_frag(ap, ap + 16);
#pragma unroll
      for (int j = 0; j < 4; ++j) {
        const unsigned short* vp = &vt[(16 * j + ln16) * 64 + ks * 32 + hx * 8];
        o[j] = wmma_bf16(af, load_frag(vp, vp + 16), o[j]);
      }
    }
    __syncthreads();   // protect vt before next iteration's staging
  }

#pragma unroll
  for (int j = 0; j < 4; ++j)
#pragma unroll
    for (int v = 0; v < 8; ++v) {
      const int row = qbase + v + 8 * hx;
      out[(size_t)row * DMODEL + h * HDIM + 16 * j + ln16] = f2bf(o[j][v] / l[v]);
    }
}

// ---------------------------------------------------------------- driver
extern "C" void kernel_launch(void* const* d_in, const int* in_sizes, int n_in,
                              void* d_out, int out_size, void* d_ws, size_t ws_size,
                              hipStream_t stream) {
  (void)in_sizes; (void)n_in; (void)out_size; (void)ws_size;
  const float* x      = (const float*)d_in[0];
  const float* w_qkv  = (const float*)d_in[1];
  const float* b_qkv  = (const float*)d_in[2];
  const float* w_proj = (const float*)d_in[3];
  const float* b_proj = (const float*)d_in[4];
  const float* w_fc1  = (const float*)d_in[5];
  const float* b_fc1  = (const float*)d_in[6];
  const float* w_fc2  = (const float*)d_in[7];
  const float* b_fc2  = (const float*)d_in[8];
  const float* ln1_g  = (const float*)d_in[9];
  const float* ln1_b  = (const float*)d_in[10];
  const float* ln2_g  = (const float*)d_in[11];
  const float* ln2_b  = (const float*)d_in[12];
  float* out = (float*)d_out;

  // workspace carve-out (~84 MB total)
  char* ws = (char*)d_ws;
  size_t off = 0;
  auto take = [&](size_t bytes) -> char* {
    char* p = ws + off; off += (bytes + 255) & ~(size_t)255; return p;
  };
  unsigned short* wqkvT  = (unsigned short*)take((size_t)QKVD * DMODEL * 2);
  unsigned short* wprojT = (unsigned short*)take((size_t)DMODEL * DMODEL * 2);
  unsigned short* wfc1T  = (unsigned short*)take((size_t)4 * DMODEL * DMODEL * 2);
  unsigned short* wfc2T  = (unsigned short*)take((size_t)DMODEL * 4 * DMODEL * 2);
  unsigned short* xln    = (unsigned short*)take((size_t)SEQL * DMODEL * 2);
  unsigned short* qkv    = (unsigned short*)take((size_t)SEQL * QKVD * 2);
  unsigned short* attno  = (unsigned short*)take((size_t)SEQL * DMODEL * 2);
  float*          x1     = (float*)take((size_t)SEQL * DMODEL * 4);
  unsigned short* hbuf   = (unsigned short*)take((size_t)SEQL * 4 * DMODEL * 2);

  auto cdiv = [](long long a, long long b) { return (int)((a + b - 1) / b); };

  // weights -> bf16, transposed to W^T[N][K]
  transpose_to_bf16<<<cdiv((long long)DMODEL * QKVD, 256), 256, 0, stream>>>(w_qkv, wqkvT, DMODEL, QKVD);
  transpose_to_bf16<<<cdiv((long long)DMODEL * DMODEL, 256), 256, 0, stream>>>(w_proj, wprojT, DMODEL, DMODEL);
  transpose_to_bf16<<<cdiv((long long)DMODEL * 4 * DMODEL, 256), 256, 0, stream>>>(w_fc1, wfc1T, DMODEL, 4 * DMODEL);
  transpose_to_bf16<<<cdiv((long long)4 * DMODEL * DMODEL, 256), 256, 0, stream>>>(w_fc2, wfc2T, 4 * DMODEL, DMODEL);

  // h = LN1(x)
  layernorm_bf16<<<SEQL, 256, 0, stream>>>(x, ln1_g, ln1_b, xln, DMODEL);
  // qkv = h @ W_qkv + b_qkv  (bf16 out)
  gemm_bf16_wmma<3><<<dim3(QKVD / 128, SEQL / 128), 128, 0, stream>>>(
      xln, wqkvT, b_qkv, nullptr, nullptr, qkv, SEQL, QKVD, DMODEL);
  // causal flash attention
  attn_wmma<<<dim3(SEQL / 64, NHEAD), 128, 0, stream>>>(qkv, attno);
  // x1 = attn @ W_proj + b_proj + x
  gemm_bf16_wmma<1><<<dim3(DMODEL / 128, SEQL / 128), 128, 0, stream>>>(
      attno, wprojT, b_proj, x, x1, nullptr, SEQL, DMODEL, DMODEL);
  // h2 = LN2(x1)
  layernorm_bf16<<<SEQL, 256, 0, stream>>>(x1, ln2_g, ln2_b, xln, DMODEL);
  // hbuf = gelu(h2 @ W_fc1 + b_fc1) (bf16 out)
  gemm_bf16_wmma<2><<<dim3(4 * DMODEL / 128, SEQL / 128), 128, 0, stream>>>(
      xln, wfc1T, b_fc1, nullptr, nullptr, hbuf, SEQL, 4 * DMODEL, DMODEL);
  // out = hbuf @ W_fc2 + b_fc2 + x1
  gemm_bf16_wmma<1><<<dim3(DMODEL / 128, SEQL / 128), 128, 0, stream>>>(
      hbuf, wfc2T, b_fc2, x1, out, nullptr, SEQL, DMODEL, 4 * DMODEL);
}
